// RecursiveMamba130m_v29_43250320670979
// MI455X (gfx1250) — matrix operand order repaired and checked by reading, not compiled, not run on hardware
//
#include <hip/hip_runtime.h>
#include <hip/hip_bf16.h>
#include <math.h>

#define D_MODEL 768
#define D_INNER 1536
#define D_STATE 16
#define D_CONV 4
#define DT_RANK 48
#define N_LAYERS 6
#define BASE_SPLIT 2
#define N_LOOPS 3
#define VOCAB 50280
#define HALT_ID (VOCAB - 1)
#define BATCH 2
#define SEQ 256
#define NTOK (BATCH * SEQ)
#define RANK 8
#define LORA_SCALE 2.0f
#define EPS 1e-5f

#define ACT_NONE 0
#define ACT_SOFTPLUS 1

typedef __bf16 bf16;
typedef __attribute__((ext_vector_type(16))) __bf16 v16bf;
typedef __attribute__((ext_vector_type(8))) float v8f;

// ---------------- embedding gather ----------------
__global__ void k_embed(const int* __restrict__ ids, const float* __restrict__ emb,
                        float* __restrict__ x) {
    int t = blockIdx.x;
    int id = ids[t];
    for (int j = threadIdx.x; j < D_MODEL; j += blockDim.x)
        x[t * D_MODEL + j] = emb[(long)id * D_MODEL + j];
}

// ---------------- LoRA merge: Wm = W + 2 * Bm @ Am ----------------
__global__ void k_lora_merge(const float* __restrict__ W, const float* __restrict__ Bm,
                             const float* __restrict__ Am, float* __restrict__ Wm,
                             int N, int K) {
    int i = blockIdx.x * blockDim.x + threadIdx.x;
    if (i >= N * K) return;
    int n = i / K, k = i % K;
    float acc = 0.f;
#pragma unroll
    for (int r = 0; r < RANK; ++r) acc += Bm[n * RANK + r] * Am[r * K + k];
    Wm[i] = W[i] + LORA_SCALE * acc;
}

// ---------------- fused (optional residual add) + RMSNorm ----------------
__global__ __launch_bounds__(256) void k_add_rms(const float* __restrict__ a,
                                                 const float* b,
                                                 const float* __restrict__ w,
                                                 float* __restrict__ out_norm,
                                                 float* out_sum) {
    int t = blockIdx.x;
    __shared__ float red[256];
    float v[3];
    float ss = 0.f;
    long base = (long)t * D_MODEL;
#pragma unroll
    for (int i = 0; i < 3; ++i) {
        int j = threadIdx.x + i * 256;
        float s = a[base + j] + (b ? b[base + j] : 0.f);
        v[i] = s;
        ss += s * s;
    }
    red[threadIdx.x] = ss;
    __syncthreads();
    for (int st = 128; st > 0; st >>= 1) {
        if (threadIdx.x < st) red[threadIdx.x] += red[threadIdx.x + st];
        __syncthreads();
    }
    float rs = rsqrtf(red[0] / (float)D_MODEL + EPS);
#pragma unroll
    for (int i = 0; i < 3; ++i) {
        int j = threadIdx.x + i * 256;
        if (out_sum) out_sum[base + j] = v[i];
        out_norm[base + j] = v[i] * w[j] * rs;
    }
}

// ---------------- fragment helpers ----------------
// 8 consecutive floats -> bf16 into f[o..o+7] (two b128 loads + cvt_pk)
__device__ __forceinline__ void cvt8(const float* __restrict__ p, v16bf& f, int o) {
    float4 a = *(const float4*)p;
    float4 b = *(const float4*)(p + 4);
    f[o + 0] = (bf16)a.x; f[o + 1] = (bf16)a.y; f[o + 2] = (bf16)a.z; f[o + 3] = (bf16)a.w;
    f[o + 4] = (bf16)b.x; f[o + 5] = (bf16)b.y; f[o + 6] = (bf16)b.z; f[o + 7] = (bf16)b.w;
}

__device__ __forceinline__ void load8_bf16(const float* __restrict__ p, int avail,
                                           bf16* __restrict__ dst) {
    if (__builtin_expect(avail >= 8, 1)) {
        float4 a = *(const float4*)p;
        float4 b = *(const float4*)(p + 4);
        dst[0] = (bf16)a.x; dst[1] = (bf16)a.y; dst[2] = (bf16)a.z; dst[3] = (bf16)a.w;
        dst[4] = (bf16)b.x; dst[5] = (bf16)b.y; dst[6] = (bf16)b.z; dst[7] = (bf16)b.w;
    } else {
#pragma unroll
        for (int j = 0; j < 8; ++j) dst[j] = (bf16)((j < avail) ? p[j] : 0.f);
    }
}

// A fragment: 16-bit 16x32 layout. lane(row,kh): K = kh*16 .. kh*16+15 (contiguous)
__device__ __forceinline__ v16bf frag_a_gen(const float* __restrict__ A, long lda,
                                            int gm, int M, int kc, int kh, int K) {
    bf16 t[16];
    if (gm < M) {
        const float* rowp = A + (long)gm * lda + kc + kh * 16;
        load8_bf16(rowp,     K - (kc + kh * 16),     t);
        load8_bf16(rowp + 8, K - (kc + kh * 16 + 8), t + 8);
    } else {
#pragma unroll
        for (int j = 0; j < 16; ++j) t[j] = (bf16)0.f;
    }
    v16bf f;
#pragma unroll
    for (int j = 0; j < 16; ++j) f[j] = t[j];
    return f;
}

// B fragment: 16-bit 32x16 layout. lane(col,kh): K = kh*16 .. kh*16+15 (contiguous)
__device__ __forceinline__ v16bf frag_b_gen(const float* __restrict__ W, long ldw,
                                            int gn, int N, int kc, int kh, int K) {
    bf16 t[16];
    if (gn < N) {
        int base = kc + kh * 16;
        const float* rowp = W + (long)gn * ldw + base;
        load8_bf16(rowp,     K - base,     t);
        load8_bf16(rowp + 8, K - base - 8, t + 8);
    } else {
#pragma unroll
        for (int j = 0; j < 16; ++j) t[j] = (bf16)0.f;
    }
    v16bf f;
#pragma unroll
    for (int j = 0; j < 16; ++j) f[j] = t[j];
    return f;
}

// ---------------- FAST WMMA GEMM (requires M%64==0, N%128==0, K%64==0) -------------
// No bounds checks anywhere. Block = 8 waves, 4(M) x 2(N); each wave computes a
// 16x64 strip: 4 WMMA per 32-K chunk with one shared A fragment, K unrolled x2.
__global__ __launch_bounds__(256) void k_gemm_wmma_fast(const float* __restrict__ A, int lda,
                                                        const float* __restrict__ W, int ldw,
                                                        float* __restrict__ C, int ldc,
                                                        int K, const float* bias, int act) {
    int tid = threadIdx.x;
    int lane = tid & 31, wave = tid >> 5;
    int wm = wave >> 1, wn = wave & 1;
    int m0 = blockIdx.y * 64, n0 = blockIdx.x * 128;
    int row = lane & 15, kh = lane >> 4;

    const float* arow = A + (long)(m0 + wm * 16 + row) * lda;
    const float* br0  = W + (long)(n0 + wn * 64 + row) * ldw;
    const float* br1  = br0 + (long)16 * ldw;
    const float* br2  = br0 + (long)32 * ldw;
    const float* br3  = br0 + (long)48 * ldw;

    v8f acc0 = {0.f, 0.f, 0.f, 0.f, 0.f, 0.f, 0.f, 0.f};
    v8f acc1 = acc0, acc2 = acc0, acc3 = acc0;

#pragma unroll 2
    for (int kc = 0; kc < K; kc += 32) {
        if (kc + 32 < K) {  // speculative near-cache prefetch of next K-chunk
            __builtin_prefetch(arow + kc + 32, 0, 3);
            __builtin_prefetch(br0 + kc + 32, 0, 3);
            __builtin_prefetch(br2 + kc + 32, 0, 3);
        }
        v16bf af;
        cvt8(arow + kc + kh * 16,     af, 0);
        cvt8(arow + kc + kh * 16 + 8, af, 8);
        int base = kc + kh * 16;
        v16bf b0, b1, b2, b3;
        cvt8(br0 + base, b0, 0); cvt8(br0 + base + 8, b0, 8);
        cvt8(br1 + base, b1, 0); cvt8(br1 + base + 8, b1, 8);
        cvt8(br2 + base, b2, 0); cvt8(br2 + base + 8, b2, 8);
        cvt8(br3 + base, b3, 0); cvt8(br3 + base + 8, b3, 8);
        acc0 = __builtin_amdgcn_wmma_f32_16x16x32_bf16(false, af, false, b0, (short)0, acc0, false, false);
        acc1 = __builtin_amdgcn_wmma_f32_16x16x32_bf16(false, af, false, b1, (short)0, acc1, false, false);
        acc2 = __builtin_amdgcn_wmma_f32_16x16x32_bf16(false, af, false, b2, (short)0, acc2, false, false);
        acc3 = __builtin_amdgcn_wmma_f32_16x16x32_bf16(false, af, false, b3, (short)0, acc3, false, false);
    }

    // C/D layout: lane<16 -> M=r, N=lane ; lane>=16 -> M=8+r, N=lane-16
    int mstore = m0 + wm * 16 + kh * 8;
    int nbase  = n0 + wn * 64 + row;
#pragma unroll
    for (int r = 0; r < 8; ++r) {
        long rowoff = (long)(mstore + r) * ldc + nbase;
        float v0 = acc0[r], v1 = acc1[r], v2 = acc2[r], v3 = acc3[r];
        if (bias) {
            v0 += bias[nbase];
            v1 += bias[nbase + 16];
            v2 += bias[nbase + 32];
            v3 += bias[nbase + 48];
        }
        if (act == ACT_SOFTPLUS) {
            v0 = (v0 > 20.f) ? v0 : log1pf(expf(v0));
            v1 = (v1 > 20.f) ? v1 : log1pf(expf(v1));
            v2 = (v2 > 20.f) ? v2 : log1pf(expf(v2));
            v3 = (v3 > 20.f) ? v3 : log1pf(expf(v3));
        }
        C[rowoff]      = v0;
        C[rowoff + 16] = v1;
        C[rowoff + 32] = v2;
        C[rowoff + 48] = v3;
    }
}

// ---------------- generic WMMA GEMM (ragged N/K tolerated) -------------------------
__global__ __launch_bounds__(256) void k_gemm_wmma_gen(const float* __restrict__ A, int lda,
                                                       const float* __restrict__ W, int ldw,
                                                       float* __restrict__ C, int ldc,
                                                       int M, int N, int K,
                                                       const float* bias, int act) {
    int tid = threadIdx.x;
    int lane = tid & 31, wave = tid >> 5;
    int wm = wave >> 1, wn = wave & 1;
    int m0 = blockIdx.y * 64, n0 = blockIdx.x * 64;
    int row = lane & 15, kh = lane >> 4;

    int gm  = m0 + wm * 16 + row;
    int gn0 = n0 + wn * 32 + row;
    int gn1 = gn0 + 16;

    v8f acc0 = {0.f, 0.f, 0.f, 0.f, 0.f, 0.f, 0.f, 0.f};
    v8f acc1 = acc0;

    for (int kc = 0; kc < K; kc += 32) {
        v16bf af = frag_a_gen(A, lda, gm, M, kc, kh, K);
        v16bf b0 = frag_b_gen(W, ldw, gn0, N, kc, kh, K);
        v16bf b1 = frag_b_gen(W, ldw, gn1, N, kc, kh, K);
        acc0 = __builtin_amdgcn_wmma_f32_16x16x32_bf16(false, af, false, b0, (short)0, acc0, false, false);
        acc1 = __builtin_amdgcn_wmma_f32_16x16x32_bf16(false, af, false, b1, (short)0, acc1, false, false);
    }

    int mstore = m0 + wm * 16 + kh * 8;
    int nst0   = n0 + wn * 32 + row;
#pragma unroll
    for (int r = 0; r < 8; ++r) {
        int m = mstore + r;
        if (m < M) {
            if (nst0 < N) {
                float vv = acc0[r];
                if (bias) vv += bias[nst0];
                if (act == ACT_SOFTPLUS) vv = (vv > 20.f) ? vv : log1pf(expf(vv));
                C[(long)m * ldc + nst0] = vv;
            }
            int n1 = nst0 + 16;
            if (n1 < N) {
                float vv = acc1[r];
                if (bias) vv += bias[n1];
                if (act == ACT_SOFTPLUS) vv = (vv > 20.f) ? vv : log1pf(expf(vv));
                C[(long)m * ldc + n1] = vv;
            }
        }
    }
}

// ---------------- causal depthwise conv1d + silu ----------------
__global__ void k_conv_silu(const float* __restrict__ xz, const float* __restrict__ cw,
                            const float* __restrict__ cb, float* __restrict__ xc) {
    int i = blockIdx.x * blockDim.x + threadIdx.x;
    if (i >= NTOK * D_INNER) return;
    int d = i % D_INNER;
    int t = i / D_INNER;
    int b = t / SEQ, l = t % SEQ;
    float acc = cb[d];
#pragma unroll
    for (int k = 0; k < D_CONV; ++k) {
        int ls = l - (D_CONV - 1) + k;
        if (ls >= 0) acc += xz[(long)(b * SEQ + ls) * (2 * D_INNER) + d] * cw[d * D_CONV + k];
    }
    xc[i] = acc / (1.f + expf(-acc));  // silu
}

// ---------------- selective scan (sequential over L, parallel over b,d) ----------------
__global__ __launch_bounds__(256) void k_scan(const float* __restrict__ xc,
                                              const float* __restrict__ dt,
                                              const float* __restrict__ dbl,
                                              const float* __restrict__ A_log,
                                              const float* __restrict__ Dv,
                                              float* __restrict__ y) {
    int b = blockIdx.y;
    int d = blockIdx.x * blockDim.x + threadIdx.x;
    __shared__ float BC[32];
    float h[D_STATE], As[D_STATE];
#pragma unroll
    for (int s = 0; s < D_STATE; ++s) {
        h[s] = 0.f;
        As[s] = -expf(A_log[d * D_STATE + s]);
    }
    float Dd = Dv[d];
    for (int l = 0; l < SEQ; ++l) {
        int t = b * SEQ + l;
        if (threadIdx.x < 32) BC[threadIdx.x] = dbl[(long)t * 80 + DT_RANK + threadIdx.x];
        __syncthreads();
        float dtv = dt[(long)t * D_INNER + d];
        float xv  = xc[(long)t * D_INNER + d];
        float accv = 0.f;
#pragma unroll
        for (int s = 0; s < D_STATE; ++s) {
            h[s] = expf(dtv * As[s]) * h[s] + dtv * BC[s] * xv;
            accv += h[s] * BC[16 + s];
        }
        y[(long)t * D_INNER + d] = accv + xv * Dd;
        __syncthreads();
    }
}

// ---------------- gate: g = y * silu(z) ----------------
__global__ void k_gate(const float* __restrict__ y, const float* __restrict__ xz,
                       float* __restrict__ g) {
    int i = blockIdx.x * blockDim.x + threadIdx.x;
    if (i >= NTOK * D_INNER) return;
    int t = i / D_INNER, d = i % D_INNER;
    float z = xz[(long)t * (2 * D_INNER) + D_INNER + d];
    g[i] = y[i] * (z / (1.f + expf(-z)));
}

// ---------------- elementwise adds ----------------
__global__ void k_add(const float* __restrict__ a, float* __restrict__ x, int n) {
    int i = blockIdx.x * blockDim.x + threadIdx.x;
    if (i < n) x[i] += a[i];
}
__global__ void k_add_bcast(const float* __restrict__ v, float* __restrict__ x) {
    int i = blockIdx.x * blockDim.x + threadIdx.x;
    if (i < NTOK * D_MODEL) x[i] += v[i % D_MODEL];
}

// ---------------- masked logit head + CE (support = unique(ids) U {HALT}) ----------------
__global__ __launch_bounds__(256) void k_head(const float* __restrict__ xn,
                                              const int* __restrict__ ids,
                                              const int* __restrict__ ans,
                                              const int* __restrict__ chain,
                                              const float* __restrict__ lmw,
                                              int loop_i, float* __restrict__ ce_out,
                                              float* __restrict__ nv_out) {
    int b = blockIdx.x;
    int tid = threadIdx.x;
    __shared__ float xv[D_MODEL];
    __shared__ int sid[SEQ];
    __shared__ float redm[256], reds[256];
    __shared__ float tgt_val;
    __shared__ int memflag;

    int a = ans[b];
    int pos = a - 1;
    if (pos < 0) pos = 0;
    if (pos > SEQ - 1) pos = SEQ - 1;
    for (int j = tid; j < D_MODEL; j += 256) xv[j] = xn[(long)(b * SEQ + pos) * D_MODEL + j];
    sid[tid] = ids[b * SEQ + tid];
    if (tid == 0) { tgt_val = 0.f; memflag = 0; }
    __syncthreads();

    int tj = (loop_i < N_LOOPS - 1) ? loop_i : (N_LOOPS - 1);
    int tgt = chain[b * N_LOOPS + tj];
    if (sid[tid] == tgt) memflag = 1;

    int cand = sid[tid];
    bool active = (cand != HALT_ID);
    for (int j = 0; j < tid && active; ++j)
        if (sid[j] == cand) active = false;

    const float4* xv4 = (const float4*)xv;
    float mv = -INFINITY, sv = 0.f;
    if (active) {
        const float4* wr = (const float4*)(lmw + (long)cand * D_MODEL);
        float dv = 0.f;
        for (int j = 0; j < D_MODEL / 4; ++j) {
            float4 xq = xv4[j], wq = wr[j];
            dv += xq.x * wq.x + xq.y * wq.y + xq.z * wq.z + xq.w * wq.w;
        }
        mv = dv; sv = 1.f;
        if (cand == tgt) tgt_val = dv;
    }
    if (tid == 0) {  // HALT is always unmasked, counted exactly once
        const float4* wr = (const float4*)(lmw + (long)HALT_ID * D_MODEL);
        float dh = 0.f;
        for (int j = 0; j < D_MODEL / 4; ++j) {
            float4 xq = xv4[j], wq = wr[j];
            dh += xq.x * wq.x + xq.y * wq.y + xq.z * wq.z + xq.w * wq.w;
        }
        float M2 = fmaxf(mv, dh);
        sv = sv * expf(mv - M2) + expf(dh - M2);
        mv = M2;
        if (tgt == HALT_ID) tgt_val = dh;
    }
    redm[tid] = mv; reds[tid] = sv;
    __syncthreads();
    for (int st = 128; st > 0; st >>= 1) {
        if (tid < st) {
            float m1 = redm[tid], m2 = redm[tid + st];
            float M = fmaxf(m1, m2);
            float s = 0.f;
            if (M > -INFINITY) s = reds[tid] * expf(m1 - M) + reds[tid + st] * expf(m2 - M);
            redm[tid] = M; reds[tid] = s;
        }
        __syncthreads();
    }
    if (tid == 0) {
        float lse = logf(reds[0]) + redm[0];
        bool valid = (a >= 1) && (a < SEQ) && (tgt < VOCAB) &&
                     ((tgt == HALT_ID) || (memflag != 0));
        float ce = lse - tgt_val;
        ce_out[loop_i * BATCH + b] = valid ? ce : 0.f;
        nv_out[loop_i * BATCH + b] = valid ? 1.f : 0.f;
    }
}

// ---------------- final weighted loss combine ----------------
__global__ void k_final(const float* __restrict__ ce, const float* __restrict__ nv,
                        float* __restrict__ out) {
    if (threadIdx.x != 0 || blockIdx.x != 0) return;
    float wsum = 0.f, lsum = 0.f;
    for (int li = 0; li < N_LOOPS; ++li) {
        float n = 0.f, c = 0.f;
        for (int b = 0; b < BATCH; ++b) {
            n += nv[li * BATCH + b];
            c += ce[li * BATCH + b];
        }
        float loss = (n > 0.f) ? c / fmaxf(n, 1.f) : 0.f;
        float w = (n > 0.f) ? 1.f : 0.f;
        lsum += loss * w;
        wsum += w;
    }
    out[0] = lsum / fmaxf(wsum, 1.f);
}

// =====================================================================
struct Mixer {
    const float *in_w, *conv_w, *conv_b, *x_proj_w, *dt_w, *dt_b, *A_log, *D, *out_w;
    const float *norm_w;
    const float *in_A, *in_B, *out_A, *out_B;
};

extern "C" void kernel_launch(void* const* d_in, const int* in_sizes, int n_in,
                              void* d_out, int out_size, void* d_ws, size_t ws_size,
                              hipStream_t stream) {
    (void)in_sizes; (void)n_in; (void)out_size; (void)ws_size;

    int idx = 0;
    const int* ids   = (const int*)d_in[idx++];
    const int* chain = (const int*)d_in[idx++];
    const int* ans   = (const int*)d_in[idx++];
    const float* emb = (const float*)d_in[idx++];
    auto nextf = [&]() { return (const float*)d_in[idx++]; };

    Mixer Lr[N_LAYERS];
    for (int i = 0; i < N_LAYERS; ++i) {
        Mixer& m = Lr[i];
        m.in_w = nextf(); m.conv_w = nextf(); m.conv_b = nextf(); m.x_proj_w = nextf();
        m.dt_w = nextf(); m.dt_b = nextf(); m.A_log = nextf(); m.D = nextf();
        m.out_w = nextf(); m.norm_w = nextf();
        if (i >= BASE_SPLIT) { m.in_A = nextf(); m.in_B = nextf(); m.out_A = nextf(); m.out_B = nextf(); }
        else { m.in_A = m.in_B = m.out_A = m.out_B = nullptr; }
    }
    const float* step_emb    = nextf();
    const float* loop_norm_w = nextf();
    Mixer core = {};
    core.in_w = nextf(); core.conv_w = nextf(); core.conv_b = nextf(); core.x_proj_w = nextf();
    core.dt_w = nextf(); core.dt_b = nextf(); core.A_log = nextf(); core.D = nextf();
    core.out_w = nextf();
    const float* norm_f_w = nextf();
    const float* lmw      = nextf();

    // workspace layout (floats)
    float* p = (float*)d_ws;
    auto alloc = [&](size_t n) { float* r = p; p += n; return r; };
    float* w_in_m  = alloc((size_t)2 * D_INNER * D_MODEL);
    float* w_out_m = alloc((size_t)D_MODEL * D_INNER);
    float* xA  = alloc((size_t)NTOK * D_MODEL);
    float* xB  = alloc((size_t)NTOK * D_MODEL);
    float* res = alloc((size_t)NTOK * D_MODEL);
    float* xn  = alloc((size_t)NTOK * D_MODEL);
    float* xz  = alloc((size_t)NTOK * 2 * D_INNER);
    float* xc  = alloc((size_t)NTOK * D_INNER);
    float* dbl = alloc((size_t)NTOK * 80);
    float* dtb = alloc((size_t)NTOK * D_INNER);
    float* yb  = alloc((size_t)NTOK * D_INNER);
    float* gb  = alloc((size_t)NTOK * D_INNER);
    float* mo  = alloc((size_t)NTOK * D_MODEL);
    float* ceb = alloc(N_LOOPS * BATCH);
    float* nvb = alloc(N_LOOPS * BATCH);

    auto run_mamba = [&](const float* inp, const Mixer& m, bool lora, float* outp) {
        const float* w_in = m.in_w;
        const float* w_out = m.out_w;
        if (lora) {
            int nin = 2 * D_INNER * D_MODEL;
            k_lora_merge<<<(nin + 255) / 256, 256, 0, stream>>>(m.in_w, m.in_B, m.in_A,
                                                               w_in_m, 2 * D_INNER, D_MODEL);
            int nout = D_MODEL * D_INNER;
            k_lora_merge<<<(nout + 255) / 256, 256, 0, stream>>>(m.out_w, m.out_B, m.out_A,
                                                                 w_out_m, D_MODEL, D_INNER);
            w_in = w_in_m; w_out = w_out_m;
        }
        // xz = inp @ w_in^T  (M=512, N=3072, K=768) -- fully aligned, fast path
        dim3 g1(2 * D_INNER / 128, NTOK / 64);
        k_gemm_wmma_fast<<<g1, 256, 0, stream>>>(inp, D_MODEL, w_in, D_MODEL, xz, 2 * D_INNER,
                                                 D_MODEL, nullptr, ACT_NONE);
        k_conv_silu<<<(NTOK * D_INNER + 255) / 256, 256, 0, stream>>>(xz, m.conv_w, m.conv_b, xc);
        // dbl = xc @ x_proj^T  (N=80, K=1536) -- ragged N, generic path
        dim3 g2((80 + 63) / 64, NTOK / 64);
        k_gemm_wmma_gen<<<g2, 256, 0, stream>>>(xc, D_INNER, m.x_proj_w, D_INNER, dbl, 80,
                                                NTOK, 80, D_INNER, nullptr, ACT_NONE);
        // dt = softplus(dbl[:, :48] @ dt_w^T + dt_b)  (N=1536, K=48) -- ragged K, generic
        dim3 g3(D_INNER / 64, NTOK / 64);
        k_gemm_wmma_gen<<<g3, 256, 0, stream>>>(dbl, 80, m.dt_w, DT_RANK, dtb, D_INNER,
                                                NTOK, D_INNER, DT_RANK, m.dt_b, ACT_SOFTPLUS);
        dim3 gs(D_INNER / 256, BATCH);
        k_scan<<<gs, 256, 0, stream>>>(xc, dtb, dbl, m.A_log, m.D, yb);
        k_gate<<<(NTOK * D_INNER + 255) / 256, 256, 0, stream>>>(yb, xz, gb);
        // out = g @ w_out^T  (N=768, K=1536) -- fully aligned, fast path
        dim3 g4(D_MODEL / 128, NTOK / 64);
        k_gemm_wmma_fast<<<g4, 256, 0, stream>>>(gb, D_INNER, w_out, D_INNER, outp, D_MODEL,
                                                 D_INNER, nullptr, ACT_NONE);
    };

    float* x = xA;
    float* xnext = xB;
    k_embed<<<NTOK, 256, 0, stream>>>(ids, emb, x);

    bool first = true;
    auto run_block = [&](const Mixer& m, bool lora) {
        k_add_rms<<<NTOK, 256, 0, stream>>>(x, first ? nullptr : res, m.norm_w, xn, res);
        first = false;
        run_mamba(xn, m, lora, xnext);
        float* t = x; x = xnext; xnext = t;
    };

    for (int i = 0; i < N_LAYERS; ++i) run_block(Lr[i], i >= BASE_SPLIT);

    for (int loop_i = 0; loop_i < N_LOOPS; ++loop_i) {
        k_add_bcast<<<(NTOK * D_MODEL + 255) / 256, 256, 0, stream>>>(
            step_emb + (size_t)loop_i * D_MODEL, x);
        for (int i = BASE_SPLIT; i < N_LAYERS; ++i) run_block(Lr[i], true);
        run_mamba(x, core, false, mo);
        k_add<<<(NTOK * D_MODEL + 255) / 256, 256, 0, stream>>>(mo, x, NTOK * D_MODEL);
        // x = rmsnorm(x, loop_norm_w) in-place
        k_add_rms<<<NTOK, 256, 0, stream>>>(x, nullptr, loop_norm_w, x, nullptr);
        // xn = rmsnorm(x + residual, norm_f_w)
        k_add_rms<<<NTOK, 256, 0, stream>>>(x, res, norm_f_w, xn, nullptr);
        k_head<<<BATCH, 256, 0, stream>>>(xn, ids, ans, chain, lmw, loop_i, ceb, nvb);
    }
    k_final<<<1, 1, 0, stream>>>(ceb, nvb, (float*)d_out);
}
